// ImageTextCrossAttention_3109556323149
// MI455X (gfx1250) — matrix operand order repaired and checked by reading, not compiled, not run on hardware
//
#include <hip/hip_runtime.h>

// ---------------------------------------------------------------------------
// Problem constants (from reference setup_inputs)
// ---------------------------------------------------------------------------
#define B_   128
#define L_   512
#define R_   512
#define DT_  300
#define DI_  1024
#define D_   256
#define KT_PAD 320          // Dt padded up to a multiple of 64 for WMMA K-loop
#define LAMBDA_S 9.0f
#define LEAKY   0.1f
#define EPS_    1e-8f

typedef _Float16 v16h __attribute__((ext_vector_type(16)));
typedef _Float16 v8h  __attribute__((ext_vector_type(8)));
typedef float    v8f  __attribute__((ext_vector_type(8)));
typedef int      v4i  __attribute__((ext_vector_type(4)));

// Address-space-typed 16B vector pointers for the async builtin.
typedef __attribute__((address_space(1))) v4i* as1_v4i_p;
typedef __attribute__((address_space(3))) v4i* as3_v4i_p;

// ---------------------------------------------------------------------------
// CDNA5 async global->LDS copy (ASYNCcnt-tracked). Fallback: sync copy.
// ---------------------------------------------------------------------------
#if __has_builtin(__builtin_amdgcn_global_load_async_to_lds_b128) && \
    __has_builtin(__builtin_amdgcn_s_wait_asynccnt)
#define HAS_ASYNC_LDS 1
#endif

__device__ __forceinline__ void g2l_16B(const _Float16* g, _Float16* l) {
#ifdef HAS_ASYNC_LDS
    // AS1 ptr from flat global addr; AS3 ptr from low 32 bits of flat LDS addr.
    __builtin_amdgcn_global_load_async_to_lds_b128(
        (as1_v4i_p)(uintptr_t)g,
        (as3_v4i_p)(uint32_t)(uintptr_t)l,
        0, 0);
#else
    *(v8h*)l = *(const v8h*)g;
#endif
}

template <int N>
__device__ __forceinline__ void wait_asynccnt() {
#ifdef HAS_ASYNC_LDS
    __builtin_amdgcn_s_wait_asynccnt(N);
#endif
}

// ---------------------------------------------------------------------------
// Wave32 reductions
// ---------------------------------------------------------------------------
__device__ __forceinline__ float wave_sum(float v) {
#pragma unroll
    for (int off = 16; off > 0; off >>= 1) v += __shfl_xor(v, off, 32);
    return v;
}
__device__ __forceinline__ float wave_max(float v) {
#pragma unroll
    for (int off = 16; off > 0; off >>= 1) v = fmaxf(v, __shfl_xor(v, off, 32));
    return v;
}

// ---------------------------------------------------------------------------
// WMMA fragment loads (wave32; contiguous k order used by this code for A and B, not the ISA table order)
// A 16x32 f16: lane<16 holds K=[0..15]; lane>=16 holds K=[16..31]
// B 32x16 f16: lane<16 holds K=[0..15]; lane>=16 holds K=[16..31] (N-major src)
// ---------------------------------------------------------------------------
__device__ __forceinline__ v16h combine16(v8h lo, v8h hi) {
    v16h r;
#pragma unroll
    for (int i = 0; i < 8; ++i) { r[i] = lo[i]; r[i + 8] = hi[i]; }
    return r;
}
__device__ __forceinline__ v16h loadA16(const _Float16* p) {   // + k0 + (lane>>4)*16
    return combine16(*(const v8h*)p, *(const v8h*)(p + 8));
}
__device__ __forceinline__ v16h loadB16(const _Float16* p) {   // + k0 + (lane>>4)*16
    return combine16(*(const v8h*)p, *(const v8h*)(p + 8));
}

// ---------------------------------------------------------------------------
// Batched f16 GEMM:  C[b] (MxN) = A[b] (MxK, row-major) * Bt[b] (NxK, K-major)^T
// block = 256 threads = 8 waves; wave computes 16(M) x 64(N); block 128 x 64.
// K stages of 64 double-buffered through LDS with async global->LDS copies.
// Per stage: hoist all 20 ds_load_b128 fragments, then 8 back-to-back WMMAs.
// Requires: M % 128 == 0, N % 64 == 0, K % 64 == 0 (true for all call sites).
// grid = (N/64, M/128, batches)
// epil 0: f32 store (+bias if non-null) -> Cf ; epil 1: leaky-relu f16 -> Ch
// ---------------------------------------------------------------------------
#define BM 128
#define BN 64
#define BK 64
#define LDSS 72   // padded f16 row stride (144B) to spread LDS banks

__global__ __launch_bounds__(256)
void wmma_gemm_f16(const _Float16* __restrict__ A, int lda, long long sA,
                   const _Float16* __restrict__ Bt, int ldb, long long sB,
                   float* __restrict__ Cf, _Float16* __restrict__ Ch,
                   int ldc, long long sC,
                   int M, int N, int K,
                   const float* __restrict__ bias, int epil)
{
    __shared__ _Float16 As[2][BM * LDSS];
    __shared__ _Float16 Bs[2][BN * LDSS];

    const int tid   = threadIdx.x;
    const int lane  = tid & 31;
    const int wave  = tid >> 5;
    const int tileM = blockIdx.y * BM;
    const int tileN = blockIdx.x * BN;

    const _Float16* Ab = A  + (size_t)blockIdx.z * sA;
    const _Float16* Bb = Bt + (size_t)blockIdx.z * sB;

    // --- stage-copy assignments (per thread: 4 A chunks + 2 B chunks of 16B)
    const int aRow = tid >> 1;             // 0..127
    const int aCol = (tid & 1) * 32;       // elements
    const int bRow = tid >> 2;             // 0..63
    const int bCol = (tid & 3) * 16;       // elements

    const _Float16* aSrc = Ab + (size_t)(tileM + aRow) * lda + aCol;
    const _Float16* bSrc = Bb + (size_t)(tileN + bRow) * ldb + bCol;

    auto fill = [&](int st, int k0) {
        const _Float16* ga = aSrc + k0;
        _Float16* la = &As[st][aRow * LDSS + aCol];
#pragma unroll
        for (int j = 0; j < 4; ++j) g2l_16B(ga + j * 8, la + j * 8);
        const _Float16* gb = bSrc + k0;
        _Float16* lb = &Bs[st][bRow * LDSS + bCol];
#pragma unroll
        for (int j = 0; j < 2; ++j) g2l_16B(gb + j * 8, lb + j * 8);
    };

    // --- WMMA fragment addressing
    const int l15   = lane & 15;
    const int aKoff = (lane >> 4) << 4;    // 0 or 16
    const int bKoff = (lane >> 4) << 4;    // 0 or 16

    v8f acc[4];
#pragma unroll
    for (int s = 0; s < 4; ++s) acc[s] = v8f{};

    const int nK = K / BK;
    fill(0, 0);                            // 6 async ops in flight

    for (int kt = 0; kt < nK; ++kt) {
        const int st = kt & 1;
        if (kt + 1 < nK) {
            fill(st ^ 1, (kt + 1) * BK);   // prefetch next stage (12 in flight)
            wait_asynccnt<6>();            // current stage landed (in-order)
        } else {
            wait_asynccnt<0>();
        }
        __syncthreads();

        // ---- hoist all LDS fragment loads for this stage (one ds clause) ----
        const _Float16* aBase = &As[st][(wave * 16 + l15) * LDSS + aKoff];
        v16h af0 = loadA16(aBase);         // kk = 0
        v16h af1 = loadA16(aBase + 32);    // kk = 32
        v16h bf0[4], bf1[4];
#pragma unroll
        for (int s = 0; s < 4; ++s) {
            const _Float16* bp = &Bs[st][(s * 16 + l15) * LDSS + bKoff];
            bf0[s] = loadB16(bp);
            bf1[s] = loadB16(bp + 32);
        }
        // ---- 8 back-to-back WMMAs (independent accumulators) ----
#pragma unroll
        for (int s = 0; s < 4; ++s)
            acc[s] = __builtin_amdgcn_wmma_f32_16x16x32_f16(
                         false, af0, false, bf0[s], (short)0, acc[s], false, false);
#pragma unroll
        for (int s = 0; s < 4; ++s)
            acc[s] = __builtin_amdgcn_wmma_f32_16x16x32_f16(
                         false, af1, false, bf1[s], (short)0, acc[s], false, false);

        __syncthreads();                   // stage consumed; safe to overwrite
    }

    // --- epilogue: C vgpr i -> row mb+i, col lane&15
    const int mb = tileM + wave * 16 + ((lane >> 4) << 3);
#pragma unroll
    for (int s = 0; s < 4; ++s) {
        const int n   = tileN + s * 16 + l15;
        const float bv = bias ? bias[n] : 0.0f;
#pragma unroll
        for (int i = 0; i < 8; ++i) {
            float v = acc[s][i] + bv;
            const size_t off = (size_t)blockIdx.z * sC + (size_t)(mb + i) * ldc + n;
            if (epil == 1) {
                v = v > 0.0f ? v : LEAKY * v;
                Ch[off] = (_Float16)v;
            } else {
                Cf[off] = v;
            }
        }
    }
    (void)M; (void)N;
}

// ---------------------------------------------------------------------------
// Elementwise helpers
// ---------------------------------------------------------------------------
__global__ void zero_out_k(float* out) {
    if (threadIdx.x < B_) out[threadIdx.x] = 0.0f;
}

// dst[r][k] = k < srcK ? (f16)src[r][k] : 0     (rows x dstK threads)
__global__ void cast_pad_f16(const float* __restrict__ src, _Float16* __restrict__ dst,
                             long long rows, int srcK, int dstK)
{
    size_t idx = (size_t)blockIdx.x * blockDim.x + threadIdx.x;
    if (idx >= (size_t)rows * dstK) return;
    int k = (int)(idx % dstK);
    size_t r = idx / dstK;
    dst[idx] = (k < srcK) ? (_Float16)src[r * srcK + k] : (_Float16)0.0f;
}

// W (K x N, f32) -> Wt (N x Kpad, f16), zero padded
__global__ void transpose_cast_w(const float* __restrict__ W, _Float16* __restrict__ Wt,
                                 int K, int N, int Kpad)
{
    size_t idx = (size_t)blockIdx.x * blockDim.x + threadIdx.x;
    if (idx >= (size_t)N * Kpad) return;
    int k = (int)(idx % Kpad);
    int n = (int)(idx / Kpad);
    Wt[idx] = (k < K) ? (_Float16)W[(size_t)k * N + n] : (_Float16)0.0f;
}

// ---------------------------------------------------------------------------
// Row L2-normalize (D_=256, 8 elems/lane, one wave per row).
// Writes Y (row-major f16) and Yt (per-batch transposed f16 [b][d][rowInBatch]).
// ---------------------------------------------------------------------------
__global__ void l2norm_rows(const float* __restrict__ X, _Float16* __restrict__ Y,
                            _Float16* __restrict__ Yt, int rowsPerBatch)
{
    const int lane = threadIdx.x & 31;
    const size_t row = (size_t)blockIdx.x * (blockDim.x >> 5) + (threadIdx.x >> 5);
    const int b = (int)(row / rowsPerBatch);
    const int r = (int)(row % rowsPerBatch);
    const float* x = X + row * D_;

    float v[D_ / 32];
    float ss = 0.0f;
#pragma unroll
    for (int i = 0; i < D_ / 32; ++i) { v[i] = x[i * 32 + lane]; ss += v[i] * v[i]; }
    ss = wave_sum(ss);
    const float scale = 1.0f / (sqrtf(ss) + EPS_);
#pragma unroll
    for (int i = 0; i < D_ / 32; ++i) {
        const int d = i * 32 + lane;
        const _Float16 h = (_Float16)(v[i] * scale);
        Y[row * D_ + d] = h;
        Yt[(size_t)b * D_ * rowsPerBatch + (size_t)d * rowsPerBatch + r] = h;
    }
}

// ---------------------------------------------------------------------------
// Softmax over last axis (R_=512): one wave per (b,l) row of A.
// ---------------------------------------------------------------------------
__global__ void softmax_rows_f16(const _Float16* __restrict__ A, _Float16* __restrict__ S)
{
    const int lane = threadIdx.x & 31;
    const size_t row = (size_t)blockIdx.x * (blockDim.x >> 5) + (threadIdx.x >> 5);
    const _Float16* a = A + row * R_;

    float v[R_ / 32];
    float mx = -3.0e38f;
#pragma unroll
    for (int i = 0; i < R_ / 32; ++i) {
        v[i] = LAMBDA_S * (float)a[i * 32 + lane];
        mx = fmaxf(mx, v[i]);
    }
    mx = wave_max(mx);
    float s = 0.0f;
#pragma unroll
    for (int i = 0; i < R_ / 32; ++i) { v[i] = __expf(v[i] - mx); s += v[i]; }
    s = wave_sum(s);
    const float inv = 1.0f / s;
#pragma unroll
    for (int i = 0; i < R_ / 32; ++i)
        S[row * R_ + i * 32 + lane] = (_Float16)(v[i] * inv);
}

// ---------------------------------------------------------------------------
// Softmax over L axis (columns of A): one wave per (b,r).
// Writes transposed result St[b][r][l] (row-major) for the following GEMM.
// ---------------------------------------------------------------------------
__global__ void softmax_cols_f16(const _Float16* __restrict__ A, _Float16* __restrict__ St)
{
    const int lane = threadIdx.x & 31;
    const size_t p = (size_t)blockIdx.x * (blockDim.x >> 5) + (threadIdx.x >> 5);
    const int b = (int)(p / R_);
    const int r = (int)(p % R_);
    const _Float16* a = A + (size_t)b * L_ * R_ + r;

    float v[L_ / 32];
    float mx = -3.0e38f;
#pragma unroll
    for (int i = 0; i < L_ / 32; ++i) {
        v[i] = LAMBDA_S * (float)a[(size_t)(i * 32 + lane) * R_];
        mx = fmaxf(mx, v[i]);
    }
    mx = wave_max(mx);
    float s = 0.0f;
#pragma unroll
    for (int i = 0; i < L_ / 32; ++i) { v[i] = __expf(v[i] - mx); s += v[i]; }
    s = wave_sum(s);
    const float inv = 1.0f / s;
    _Float16* o = St + (size_t)b * R_ * L_ + (size_t)r * L_;
#pragma unroll
    for (int i = 0; i < L_ / 32; ++i)
        o[i * 32 + lane] = (_Float16)(v[i] * inv);
}

// ---------------------------------------------------------------------------
// Mean cosine per batch: one wave per row; atomicAdd(cos * invCount) into out[b].
// ---------------------------------------------------------------------------
__global__ void cosine_mean_atomic(const _Float16* __restrict__ Xh, const float* __restrict__ Y,
                                   float* __restrict__ out, int rowsPerBatch, float invCount)
{
    const int lane = threadIdx.x & 31;
    const size_t row = (size_t)blockIdx.x * (blockDim.x >> 5) + (threadIdx.x >> 5);
    const int b = (int)(row / rowsPerBatch);

    float dot = 0.0f, nx = 0.0f, ny = 0.0f;
#pragma unroll
    for (int i = 0; i < D_ / 32; ++i) {
        const size_t idx = row * D_ + i * 32 + lane;
        const float xv = (float)Xh[idx];
        const float yv = Y[idx];
        dot += xv * yv; nx += xv * xv; ny += yv * yv;
    }
    dot = wave_sum(dot); nx = wave_sum(nx); ny = wave_sum(ny);
    if (lane == 0) {
        const float c = dot / (sqrtf(nx) * sqrtf(ny) + EPS_);
        atomicAdd(&out[b], c * invCount);
    }
}

// ---------------------------------------------------------------------------
// Launch
// ---------------------------------------------------------------------------
extern "C" void kernel_launch(void* const* d_in, const int* in_sizes, int n_in,
                              void* d_out, int out_size, void* d_ws, size_t ws_size,
                              hipStream_t stream)
{
    const float* text  = (const float*)d_in[0];
    const float* image = (const float*)d_in[1];
    const float* Wt    = (const float*)d_in[2];
    const float* bt    = (const float*)d_in[3];
    const float* Wi    = (const float*)d_in[4];
    const float* bi    = (const float*)d_in[5];
    float* out = (float*)d_out;

    char* w = (char*)d_ws;
    size_t off = 0;
    auto take = [&](size_t bytes) -> char* {
        char* p = w + off;
        off = (off + bytes + 255) & ~(size_t)255;
        return p;
    };
    const size_t BL = (size_t)B_ * L_;   // 65536
    const size_t BR = (size_t)B_ * R_;   // 65536

    _Float16* textH  = (_Float16*)take(BL * KT_PAD * 2);
    _Float16* imageH = (_Float16*)take(BR * DI_ * 2);
    _Float16* WtT    = (_Float16*)take((size_t)D_ * KT_PAD * 2);
    _Float16* WiT    = (_Float16*)take((size_t)D_ * DI_ * 2);
    float*    tfRaw  = (float*)take(BL * D_ * 4);       // later reused as image_ss
    float*    imfRaw = (float*)take(BR * D_ * 4);       // later reused as text_ss
    _Float16* tfH    = (_Float16*)take(BL * D_ * 2);
    _Float16* imfH   = (_Float16*)take(BR * D_ * 2);
    _Float16* tfT    = (_Float16*)take(BL * D_ * 2);    // [b][d][l]
    _Float16* imfT   = (_Float16*)take(BR * D_ * 2);    // [b][d][r]
    _Float16* Ah     = (_Float16*)take(BL * R_ * 2);    // leaky-relu'd scores
    _Float16* Stti   = (_Float16*)take(BL * R_ * 2);
    _Float16* SittT  = (_Float16*)take(BR * L_ * 2);    // [b][r][l]

    zero_out_k<<<1, 128, 0, stream>>>(out);

    { size_t n = BL * KT_PAD;
      cast_pad_f16<<<dim3((unsigned)((n + 255) / 256)), 256, 0, stream>>>(text, textH, (long long)BL, DT_, KT_PAD); }
    { size_t n = BR * DI_;
      cast_pad_f16<<<dim3((unsigned)((n + 255) / 256)), 256, 0, stream>>>(image, imageH, (long long)BR, DI_, DI_); }
    transpose_cast_w<<<dim3((unsigned)(((size_t)D_ * KT_PAD + 255) / 256)), 256, 0, stream>>>(Wt, WtT, DT_, D_, KT_PAD);
    transpose_cast_w<<<dim3((unsigned)(((size_t)D_ * DI_ + 255) / 256)), 256, 0, stream>>>(Wi, WiT, DI_, D_, DI_);

    // tf_raw = textH @ Wt + bt      (M=BL, N=256, K=320)
    wmma_gemm_f16<<<dim3(D_ / 64, (unsigned)(BL / 128), 1), 256, 0, stream>>>(
        textH, KT_PAD, 0, WtT, KT_PAD, 0, tfRaw, nullptr, D_, 0,
        (int)BL, D_, KT_PAD, bt, 0);
    // imf_raw = imageH @ Wi + bi    (M=BR, N=256, K=1024)
    wmma_gemm_f16<<<dim3(D_ / 64, (unsigned)(BR / 128), 1), 256, 0, stream>>>(
        imageH, DI_, 0, WiT, DI_, 0, imfRaw, nullptr, D_, 0,
        (int)BR, D_, DI_, bi, 0);

    l2norm_rows<<<dim3((unsigned)(BL / 8)), 256, 0, stream>>>(tfRaw, tfH, tfT, L_);
    l2norm_rows<<<dim3((unsigned)(BR / 8)), 256, 0, stream>>>(imfRaw, imfH, imfT, R_);

    // A[b] = leaky_relu(tf[b] @ imf[b]^T)   (batched: M=L, N=R, K=D)
    wmma_gemm_f16<<<dim3(R_ / 64, L_ / 128, B_), 256, 0, stream>>>(
        tfH, D_, (long long)L_ * D_, imfH, D_, (long long)R_ * D_,
        nullptr, Ah, R_, (long long)L_ * R_,
        L_, R_, D_, nullptr, 1);

    softmax_rows_f16<<<dim3((unsigned)(BL / 8)), 256, 0, stream>>>(Ah, Stti);
    softmax_cols_f16<<<dim3((unsigned)(BR / 8)), 256, 0, stream>>>(Ah, SittT);

    // image_ss[b] = S_tti[b] @ imf[b]       (M=L, N=D, K=R; B-op = imfT)
    float* imageSS = tfRaw;
    wmma_gemm_f16<<<dim3(D_ / 64, L_ / 128, B_), 256, 0, stream>>>(
        Stti, R_, (long long)L_ * R_, imfT, R_, (long long)D_ * R_,
        imageSS, nullptr, D_, (long long)L_ * D_,
        L_, D_, R_, nullptr, 0);
    // text_ss[b] = S_itt[b]^T @ tf[b]       (M=R, N=D, K=L; A-op = SittT, B-op = tfT)
    float* textSS = imfRaw;
    wmma_gemm_f16<<<dim3(D_ / 64, R_ / 128, B_), 256, 0, stream>>>(
        SittT, L_, (long long)R_ * L_, tfT, L_, (long long)D_ * L_,
        textSS, nullptr, D_, (long long)R_ * D_,
        R_, D_, L_, nullptr, 0);

    cosine_mean_atomic<<<dim3((unsigned)(BL / 8)), 256, 0, stream>>>(tfH, imageSS, out, L_, 1.0f / L_);
    cosine_mean_atomic<<<dim3((unsigned)(BR / 8)), 256, 0, stream>>>(imfH, textSS, out, R_, 1.0f / R_);

    (void)in_sizes; (void)n_in; (void)out_size; (void)ws_size;
}
